// BiMambaEncoderLayer_38233798869638
// MI455X (gfx1250) — compile-verified, matrix operand order checked by reading, not run on hardware
//
#include <hip/hip_runtime.h>
#include <cmath>

// MI455X / gfx1250: wave32, WMMA f32_16x16x32_f16.
// BiMamba layer: B=2, L=2048, D=512, ED=1024, N=16, DT_RANK=32, D_FF=1024.
//
// All GEMM operands are pre-converted to f16 (weights additionally transposed
// to [N x K]) so the WMMA inner loop is pure b128 loads + v_wmma, no cvt.

typedef __attribute__((ext_vector_type(16))) _Float16 v16h;
typedef __attribute__((ext_vector_type(8)))  _Float16 h8;
typedef __attribute__((ext_vector_type(8)))  float    v8f;

#define BB 2
#define LL 2048
#define DM 512
#define EDx 1024
#define NSTATE 16
#define DTR 32
#define DFF 1024
#define MROWS (BB * LL)   // 4096

#define FLAG_ACC  1
#define FLAG_FLIP 2
#define ACT_NONE     0
#define ACT_RELU     1
#define ACT_SOFTPLUS 2

// ---------------------------------------------------------------------------
// WMMA GEMM: act(A[M x K] * W[K x N] + bias) (+ residual) -> C (f32), Ch (f16)
//   A  : f16 row-major, leading dim lda
//   Bt : f16 W transposed, [N x K] row-major (K contiguous per output column)
// Each wave computes a 64(M) x 16*NB(N) tile: NB B-fragments + 4 A-fragments
// per 32-wide K-step feed 4*NB WMMAs (NB=2 -> 1.5 b128 loads per WMMA).
// 4 waves / block -> block tile 64 x (64*NB).
// Fragment layouts per ISA 7.12.2 (16-bit A 16x32 / B 32x16; f32 C/D 8 VGPR).
// ---------------------------------------------------------------------------
template <int NB>
__global__ void __launch_bounds__(128)
bimamba_wmma_gemm(const _Float16* __restrict__ A, const _Float16* __restrict__ Bt,
                  float* __restrict__ C, _Float16* __restrict__ Ch,
                  const float* __restrict__ bias,
                  const float* __restrict__ residual,
                  int K, int Nout, int lda, int act, int flags)
{
  const int lane    = threadIdx.x & 31;
  const int wave    = threadIdx.x >> 5;
  const int rowBase = blockIdx.y * 64;
  const int colBase = blockIdx.x * (64 * NB) + wave * (16 * NB);

  // Per-subtile A row pointers (with optional time-flip for backward mamba).
  const _Float16* arow[4];
#pragma unroll
  for (int s = 0; s < 4; ++s) {
    int mA = rowBase + 16 * s + (lane & 15);
    if (flags & FLAG_FLIP) {
      int b = mA / LL, l = mA % LL;
      mA = b * LL + (LL - 1 - l);
    }
    arow[s] = A + (size_t)mA * lda;
  }
  const int khalf = (lane < 16) ? 0 : 8;
  const _Float16* brow[NB];
#pragma unroll
  for (int nb = 0; nb < NB; ++nb)
    brow[nb] = Bt + (size_t)(colBase + 16 * nb + (lane & 15)) * K;

  v8f acc[4][NB];
#pragma unroll
  for (int s = 0; s < 4; ++s)
#pragma unroll
    for (int nb = 0; nb < NB; ++nb) acc[s][nb] = (v8f){};

  for (int k0 = 0; k0 < K; k0 += 32) {
    // Speculative prefetch of the next K panel (safe past end: ISA 10.5).
    __builtin_prefetch(brow[0] + k0 + 32, 0, 3);
    // Issue ALL fragment loads first so waits can stagger across the WMMAs.
    v16h bf[NB], af[4];
#pragma unroll
    for (int nb = 0; nb < NB; ++nb) {
      h8 lo = *(const h8*)(brow[nb] + k0 + khalf);
      h8 hi = *(const h8*)(brow[nb] + k0 + 16 + khalf);
#pragma unroll
      for (int i = 0; i < 8; ++i) { bf[nb][i] = lo[i]; bf[nb][i + 8] = hi[i]; }
    }
#pragma unroll
    for (int s = 0; s < 4; ++s) {
      h8 lo = *(const h8*)(arow[s] + k0 + khalf);
      h8 hi = *(const h8*)(arow[s] + k0 + 16 + khalf);
#pragma unroll
      for (int i = 0; i < 8; ++i) { af[s][i] = lo[i]; af[s][i + 8] = hi[i]; }
    }
#pragma unroll
    for (int s = 0; s < 4; ++s)
#pragma unroll
      for (int nb = 0; nb < NB; ++nb)
        acc[s][nb] = __builtin_amdgcn_wmma_f32_16x16x32_f16(
            false, af[s], false, bf[nb], (short)0, acc[s][nb], false, false);
  }

  // C/D layout: VGPR r -> (M = r + 8*(lane>=16), N = lane%16)
  const int mo = (lane < 16) ? 0 : 8;
  const int no = lane & 15;
#pragma unroll
  for (int s = 0; s < 4; ++s) {
#pragma unroll
    for (int nb = 0; nb < NB; ++nb) {
#pragma unroll
      for (int r = 0; r < 8; ++r) {
        int    row = rowBase + 16 * s + r + mo;
        int    col = colBase + 16 * nb + no;
        float  v   = acc[s][nb][r];
        if (bias) v += bias[col];
        if (act == ACT_RELU)          v = fmaxf(v, 0.0f);
        else if (act == ACT_SOFTPLUS) v = (v > 20.0f) ? v
                                          : __logf(1.0f + __expf(v));
        size_t o = (size_t)row * Nout + col;
        if (residual) v += residual[o];
        if (C)  { if (flags & FLAG_ACC) C[o] += v; else C[o] = v; }
        if (Ch) Ch[o] = (_Float16)v;
      }
    }
  }
}

// ---------------------------------------------------------------------------
// Weight prep: f32 [K x N] -> f16 transposed [N x K].
// ---------------------------------------------------------------------------
__global__ void __launch_bounds__(256)
bimamba_w2h_tr(const float* __restrict__ W, _Float16* __restrict__ Wt,
               int K, int N)
{
  int idx = blockIdx.x * blockDim.x + threadIdx.x;
  if (idx >= K * N) return;
  int k = idx / N, n = idx - k * N;
  Wt[(size_t)n * K + k] = (_Float16)W[idx];
}

// Activation prep: f32 -> f16 (no transpose), used for the input x.
__global__ void __launch_bounds__(256)
bimamba_f2h(const float* __restrict__ src, _Float16* __restrict__ dst, int n)
{
  int idx = blockIdx.x * blockDim.x + threadIdx.x;
  if (idx < n) dst[idx] = (_Float16)src[idx];
}

// ---------------------------------------------------------------------------
// Depthwise causal conv (D_CONV=4) + bias + SiLU over the xs half of xz.
// Writes f32 (for the scan) and f16 (A-operand of the x-proj GEMM).
// ---------------------------------------------------------------------------
__global__ void __launch_bounds__(256)
bimamba_conv_silu(const float* __restrict__ xz, const float* __restrict__ cw,
                  const float* __restrict__ cb, float* __restrict__ xc,
                  _Float16* __restrict__ xch)
{
  int idx = blockIdx.x * blockDim.x + threadIdx.x;
  if (idx >= MROWS * EDx) return;
  int e = idx & (EDx - 1);
  int m = idx >> 10;
  int b = m / LL, l = m % LL;
  float w0 = cw[e * 4 + 0], w1 = cw[e * 4 + 1];
  float w2 = cw[e * 4 + 2], w3 = cw[e * 4 + 3];
  float s = cb[e];
  int t;
  t = l - 3; if (t >= 0) s += w0 * xz[(size_t)(b * LL + t) * 2048 + e];
  t = l - 2; if (t >= 0) s += w1 * xz[(size_t)(b * LL + t) * 2048 + e];
  t = l - 1; if (t >= 0) s += w2 * xz[(size_t)(b * LL + t) * 2048 + e];
  s += w3 * xz[(size_t)m * 2048 + e];
  s = s / (1.0f + __expf(-s));            // silu
  xc[idx]  = s;
  xch[idx] = (_Float16)s;
}

// ---------------------------------------------------------------------------
// Selective scan. One lane per (b, e, n): n = tid&15. 16-lane shfl_xor tree
// reduces over the N states; lane n==0 fuses D-skip + z-gate SiLU, stores f16
// (only the out-proj GEMM consumes ys).
// ---------------------------------------------------------------------------
__global__ void __launch_bounds__(256)
bimamba_scan(const float* __restrict__ xc, const float* __restrict__ delta,
             const float* __restrict__ dbl, const float* __restrict__ xz,
             const float* __restrict__ A_log, const float* __restrict__ Dp,
             _Float16* __restrict__ ysh)
{
  int n  = threadIdx.x & 15;
  int ch = blockIdx.x * 16 + (threadIdx.x >> 4);   // (b,e) channel in [0, 2048)
  int b  = ch >> 10;
  int e  = ch & 1023;
  float Aval = -__expf(A_log[e * NSTATE + n]);
  float Dval = Dp[e];
  float h = 0.0f;
  for (int l = 0; l < LL; ++l) {
    size_t m  = (size_t)b * LL + l;
    float  dv = delta[m * EDx + e];
    float  xv = xc[m * EDx + e];
    float  Bn = dbl[m * 64 + DTR + n];
    float  Cn = dbl[m * 64 + DTR + NSTATE + n];
    h = __expf(dv * Aval) * h + (dv * xv) * Bn;
    float s = h * Cn;
    s += __shfl_xor(s, 1, 32);
    s += __shfl_xor(s, 2, 32);
    s += __shfl_xor(s, 4, 32);
    s += __shfl_xor(s, 8, 32);
    if (n == 0) {
      float y  = s + xv * Dval;
      float zv = xz[m * 2048 + EDx + e];
      y *= zv / (1.0f + __expf(-zv));     // gate by silu(z)
      ysh[m * EDx + e] = (_Float16)y;
    }
  }
}

// ---------------------------------------------------------------------------
// In-place LayerNorm over rows of 512; refreshes the f16 copy.
// ---------------------------------------------------------------------------
__global__ void __launch_bounds__(128)
bimamba_layernorm(float* __restrict__ hbuf, _Float16* __restrict__ hh,
                  const float* __restrict__ g, const float* __restrict__ bta)
{
  __shared__ float s1[128], s2[128];
  size_t base = (size_t)blockIdx.x * DM;
  float* p = hbuf + base;
  float v[4], su = 0.0f, sq = 0.0f;
#pragma unroll
  for (int i = 0; i < 4; ++i) {
    v[i] = p[threadIdx.x + 128 * i];
    su += v[i]; sq += v[i] * v[i];
  }
  s1[threadIdx.x] = su; s2[threadIdx.x] = sq;
  __syncthreads();
  for (int off = 64; off > 0; off >>= 1) {
    if ((int)threadIdx.x < off) {
      s1[threadIdx.x] += s1[threadIdx.x + off];
      s2[threadIdx.x] += s2[threadIdx.x + off];
    }
    __syncthreads();
  }
  float mu  = s1[0] * (1.0f / DM);
  float var = s2[0] * (1.0f / DM) - mu * mu;
  float inv = rsqrtf(var + 1e-5f);
#pragma unroll
  for (int i = 0; i < 4; ++i) {
    int c = threadIdx.x + 128 * i;
    float w = (v[i] - mu) * inv * g[c] + bta[c];
    p[c] = w;
    hh[base + c] = (_Float16)w;
  }
}

// ---------------------------------------------------------------------------
extern "C" void kernel_launch(void* const* d_in, const int* in_sizes, int n_in,
                              void* d_out, int out_size, void* d_ws, size_t ws_size,
                              hipStream_t stream)
{
  const float* x       = (const float*)d_in[0];
  const float* norm1_g = (const float*)d_in[19];
  const float* norm1_b = (const float*)d_in[20];
  const float* ffn_w1  = (const float*)d_in[23];
  const float* ffn_b1  = (const float*)d_in[24];
  const float* ffn_w2  = (const float*)d_in[25];
  const float* ffn_b2  = (const float*)d_in[26];

  // Workspace carve-up (64B aligned so h8/b128 loads stay 16B aligned).
  char* wsb = (char*)d_ws;
  size_t off = 0;
  auto allocF = [&](size_t n) {
    float* p = (float*)(wsb + off);
    off = (off + n * sizeof(float) + 63) & ~(size_t)63;
    return p;
  };
  auto allocH = [&](size_t n) {
    _Float16* p = (_Float16*)(wsb + off);
    off = (off + n * sizeof(_Float16) + 63) & ~(size_t)63;
    return p;
  };

  float*    xz     = allocF((size_t)MROWS * 2048); // in-proj out (xs | z)
  float*    xc     = allocF((size_t)MROWS * EDx);  // conv+silu (f32, scan)
  _Float16* xch    = allocH((size_t)MROWS * EDx);  // conv+silu (f16, GEMM A)
  float*    dbl    = allocF((size_t)MROWS * 64);   // x-proj out (dt|B|C)
  _Float16* dblh   = allocH((size_t)MROWS * 64);
  float*    dl     = allocF((size_t)MROWS * EDx);  // softplus(delta)
  _Float16* ysh    = allocH((size_t)MROWS * EDx);  // scan out (gated, f16)
  float*    mbuf   = allocF((size_t)MROWS * DM);   // mamba out / mf (f32)
  _Float16* mbufh  = allocH((size_t)MROWS * DM);
  _Float16* h1h    = allocH((size_t)MROWS * DFF);  // ffn hidden (f16 only)
  _Float16* xh     = allocH((size_t)MROWS * DM);   // x in f16
  _Float16* inw_t  = allocH((size_t)DM * 2 * EDx); // transposed f16 weights
  _Float16* xpw_t  = allocH((size_t)EDx * 64);
  _Float16* dtw_t  = allocH((size_t)DTR * EDx);
  _Float16* outw_t = allocH((size_t)EDx * DM);
  _Float16* fw1_t  = allocH((size_t)DM * DFF);
  _Float16* fw2_t  = allocH((size_t)DFF * DM);
  (void)ws_size;

  // One-time prep: x -> f16, FFN weights -> f16 transposed.
  bimamba_f2h<<<(MROWS * DM) / 256, 256, 0, stream>>>(x, xh, MROWS * DM);
  bimamba_w2h_tr<<<(DM * DFF) / 256, 256, 0, stream>>>(ffn_w1, fw1_t, DM, DFF);
  bimamba_w2h_tr<<<(DFF * DM) / 256, 256, 0, stream>>>(ffn_w2, fw2_t, DFF, DM);

  for (int dir = 0; dir < 2; ++dir) {
    const float* in_w    = (const float*)d_in[1 + dir * 9 + 0];
    const float* conv_w  = (const float*)d_in[1 + dir * 9 + 1];
    const float* conv_b  = (const float*)d_in[1 + dir * 9 + 2];
    const float* xproj_w = (const float*)d_in[1 + dir * 9 + 3];
    const float* dt_w    = (const float*)d_in[1 + dir * 9 + 4];
    const float* dt_b    = (const float*)d_in[1 + dir * 9 + 5];
    const float* A_log   = (const float*)d_in[1 + dir * 9 + 6];
    const float* Dp      = (const float*)d_in[1 + dir * 9 + 7];
    const float* out_w   = (const float*)d_in[1 + dir * 9 + 8];

    // Per-direction weight prep (buffers reused across directions).
    bimamba_w2h_tr<<<(DM * 2 * EDx) / 256, 256, 0, stream>>>(in_w, inw_t, DM, 2 * EDx);
    bimamba_w2h_tr<<<(EDx * 64) / 256, 256, 0, stream>>>(xproj_w, xpw_t, EDx, 64);
    bimamba_w2h_tr<<<(DTR * EDx) / 256, 256, 0, stream>>>(dt_w, dtw_t, DTR, EDx);
    bimamba_w2h_tr<<<(EDx * DM) / 256, 256, 0, stream>>>(out_w, outw_t, EDx, DM);

    // 1) xz = (flip?)x @ in_w           [4096x512]x[512x2048]
    bimamba_wmma_gemm<2><<<dim3(2048 / 128, MROWS / 64), 128, 0, stream>>>(
        xh, inw_t, xz, nullptr, nullptr, nullptr,
        DM, 2 * EDx, DM, ACT_NONE, dir ? FLAG_FLIP : 0);
    // 2) depthwise causal conv + silu
    bimamba_conv_silu<<<(MROWS * EDx) / 256, 256, 0, stream>>>(
        xz, conv_w, conv_b, xc, xch);
    // 3) dbl = xc @ xproj_w             [4096x1024]x[1024x64]  (narrow N)
    bimamba_wmma_gemm<1><<<dim3(64 / 64, MROWS / 64), 128, 0, stream>>>(
        xch, xpw_t, dbl, dblh, nullptr, nullptr,
        EDx, 64, EDx, ACT_NONE, 0);
    // 4) delta = softplus(dt @ dt_w + dt_b)   [4096x32]x[32x1024]
    bimamba_wmma_gemm<2><<<dim3(EDx / 128, MROWS / 64), 128, 0, stream>>>(
        dblh, dtw_t, dl, nullptr, dt_b, nullptr,
        DTR, EDx, 64, ACT_SOFTPLUS, 0);
    // 5) selective scan + D-skip + silu(z) gate
    bimamba_scan<<<(BB * EDx) / 16, 256, 0, stream>>>(
        xc, dl, dbl, xz, A_log, Dp, ysh);
    // 6) m = ys @ out_w                 [4096x1024]x[1024x512]
    bimamba_wmma_gemm<2><<<dim3(DM / 128, MROWS / 64), 128, 0, stream>>>(
        ysh, outw_t, mbuf, mbufh, nullptr, nullptr,
        EDx, DM, EDx, ACT_NONE, 0);
    // 7) layernorm (forward branch only)
    if (dir == 0)
      bimamba_layernorm<<<MROWS, 128, 0, stream>>>(mbuf, mbufh, norm1_g, norm1_b);
    // 8) h1 = relu(m @ w1 + b1)  (f16 output only; feeds ffn2 as A)
    bimamba_wmma_gemm<2><<<dim3(DFF / 128, MROWS / 64), 128, 0, stream>>>(
        mbufh, fw1_t, nullptr, h1h, ffn_b1, nullptr,
        DM, DFF, DM, ACT_RELU, 0);
    // 9) out (+)= h1 @ w2 + b2 + m   (forward writes, backward accumulates)
    bimamba_wmma_gemm<2><<<dim3(DM / 128, MROWS / 64), 128, 0, stream>>>(
        h1h, fw2_t, (float*)d_out, nullptr, ffn_b2, mbuf,
        DFF, DM, DFF, ACT_NONE, dir ? FLAG_ACC : 0);
  }
  (void)in_sizes; (void)n_in; (void)out_size;
}
